// MLSHAgent_90443421319805
// MI455X (gfx1250) — matrix operand order repaired and checked by reading, not compiled, not run on hardware
//
#include <hip/hip_runtime.h>
#include <hip/hip_bf16.h>

// MI455X (gfx1250, wave32) fused MLSH forward.
//  Pass 1 (tiny prep kernels): convert the 4 weight matrices (W_ih, W_hh,
//    Wa1, Wc1; 4 MB f32 total) into f16 WMMA B-fragment order in d_ws. They
//    are reused by all 4096 row-tile blocks, so per-block conversion (prev
//    round: ~3000 v_cvt + strided gathers per block) is wasted work.
//  Pass 2 (fused kernel): LSTM gates via v_wmma_f32_16x16x32_f16 with f32
//    accum -> pointwise epilogue in registers (hardware v_tanh/v_exp) ->
//    h_new staged f16 in LDS -> all-expert layer-1 WMMA GEMMs -> per-row
//    expert-selected layer-2 GEMVs from LDS. Gate biases arrive in LDS via
//    global_load_async_to_lds_b128 (ASYNCcnt), overlapped with the GEMM.
// B=65536, E=16, D=256, H=64, A=18. One block = 8 waves = 16 batch rows.

typedef __attribute__((ext_vector_type(16))) _Float16 v16h;
typedef __attribute__((ext_vector_type(8)))  float    v8f;
typedef __attribute__((ext_vector_type(4)))  float    v4f;

#define DD 256
#define HH 64
#define AA 18
#define EE 16
#define ACT_STRIDE (1024 + 8)  // padded f16 row stride for activations
#define HF_STRIDE  (256 + 8)   // padded f16 row stride for h_new tile

// Packed-fragment geometry: 64 column tiles x 8 k-steps x 32 lanes x 16 f16.
#define FRAG_HALVES (64 * 8 * 32 * 16)  // 262144 halves = 512 KB per matrix

// ---------------- fast math (gfx1250 hardware transcendentals) -------------
__device__ __forceinline__ float fast_tanh(float x) {
#if __has_builtin(__builtin_amdgcn_tanhf)
  return __builtin_amdgcn_tanhf(x);
#elif __has_builtin(__builtin_amdgcn_tanh_f32)
  return __builtin_amdgcn_tanh_f32(x);
#else
  return tanhf(x);
#endif
}
__device__ __forceinline__ float fast_sigmoid(float x) {
#if __has_builtin(__builtin_amdgcn_exp2f) && __has_builtin(__builtin_amdgcn_rcpf)
  return __builtin_amdgcn_rcpf(1.0f + __builtin_amdgcn_exp2f(-1.442695041f * x));
#else
  return 1.0f / (1.0f + __expf(-x));
#endif
}

// ---------------- async global->LDS staging (CDNA5, ASYNCcnt) --------------
__device__ __forceinline__ void async_cp_b128(unsigned lds_off, const void* g) {
  asm volatile("global_load_async_to_lds_b128 %0, %1, off"
               :: "v"(lds_off), "v"(g) : "memory");
}
__device__ __forceinline__ void wait_async0() {
  asm volatile("s_wait_asynccnt 0" ::: "memory");
}
__device__ __forceinline__ unsigned lds_off_of(const void* p) {
  // Flat LDS aperture addresses carry the wave-relative byte offset in [31:0].
  return (unsigned)(uintptr_t)p;
}

// ---------------- WMMA ------------------------------------------------------
__device__ __forceinline__ v8f wmma16(v16h a, v16h b, v8f c) {
  return __builtin_amdgcn_wmma_f32_16x16x32_f16(
      false, a, false, b, (short)0, c, false, false);
}

// A-fragment (16x32 f16) from row-major f32 [rows, ld].
// Lanes 0-15: rows M, K={0..7,16..23}; lanes 16-31: K={8..15,24..31}.
__device__ __forceinline__ v16h load_a_f32(const float* __restrict__ src,
                                           int ld, int row0, int k0, int lane) {
  int m = lane & 15, hi = lane >> 4;
  const float* p = src + (size_t)(row0 + m) * ld + k0 + hi * 8;
  v4f f0 = *(const v4f*)(p);
  v4f f1 = *(const v4f*)(p + 4);
  v4f f2 = *(const v4f*)(p + 16);
  v4f f3 = *(const v4f*)(p + 20);
  v16h a;
#pragma unroll
  for (int j = 0; j < 4; ++j) {
    a[j]      = (_Float16)f0[j];
    a[4 + j]  = (_Float16)f1[j];
    a[8 + j]  = (_Float16)f2[j];
    a[12 + j] = (_Float16)f3[j];
  }
  return a;
}

// A-fragment from the f16 h_new tile staged in LDS.
__device__ __forceinline__ v16h load_a_lds(const _Float16* __restrict__ hf,
                                           int k0, int lane) {
  int m = lane & 15, hi = lane >> 4;
  const _Float16* p = hf + (size_t)m * HF_STRIDE + k0 + hi * 8;
  v16h a;
#pragma unroll
  for (int j = 0; j < 8; ++j) {
    a[j]     = p[j];
    a[8 + j] = p[16 + j];
  }
  return a;
}

// Packed B-fragment: one 32-byte coalesced vector load per lane.
__device__ __forceinline__ v16h load_b_packed(const v16h* __restrict__ P,
                                              int ct, int ks, int lane) {
  return P[(ct * 8 + ks) * 32 + lane];
}

// ---------------- prep kernels: f32 -> packed f16 B fragments ---------------
// dst[((ct*8+ks)*32 + l)*16 + j] holds B(k,n) for the WMMA 32x16 f16 layout:
//   n = ct*16 + (l&15), k = ks*32 + (j>>3)*16 + (l>>4)*8 + (j&7).

// W stored [out_cols, K] row-major (LSTM W_ih/W_hh: [1024,256]; B = W^T).
__global__ __launch_bounds__(256) void pack_wT_kernel(
    const float* __restrict__ W, _Float16* __restrict__ dst) {
  int idx = blockIdx.x * 256 + threadIdx.x;  // 0..FRAG_HALVES-1
  int j = idx & 15, l = (idx >> 4) & 31, ks = (idx >> 9) & 7, ct = idx >> 12;
  int col = ct * 16 + (l & 15);
  int k   = ks * 32 + (j >> 3) * 16 + (l >> 4) * 8 + (j & 7);
  dst[idx] = (_Float16)W[(size_t)col * DD + k];
}

// W stored per-expert [K, H] row-major (Wa1/Wc1: [E,256,64]; flattened cols).
__global__ __launch_bounds__(256) void pack_wK_kernel(
    const float* __restrict__ W, _Float16* __restrict__ dst) {
  int idx = blockIdx.x * 256 + threadIdx.x;
  int j = idx & 15, l = (idx >> 4) & 31, ks = (idx >> 9) & 7, ct = idx >> 12;
  int col = ct * 16 + (l & 15);            // 0..1023 = e*64 + n
  int e = col >> 6, n = col & 63;
  int k = ks * 32 + (j >> 3) * 16 + (l >> 4) * 8 + (j & 7);
  dst[idx] = (_Float16)W[((size_t)e * DD + k) * HH + n];
}

// ---------------- fused main kernel ----------------------------------------
__global__ __launch_bounds__(256) void mlsh_fused_kernel(
    const float* __restrict__ obs, const float* __restrict__ h,
    const float* __restrict__ c, const int* __restrict__ idxs,
    const v16h* __restrict__ pWih, const v16h* __restrict__ pWhh,
    const v16h* __restrict__ pWa1, const v16h* __restrict__ pWc1,
    const float* __restrict__ b_ih, const float* __restrict__ b_hh,
    const float* __restrict__ ba1, const float* __restrict__ Wa2,
    const float* __restrict__ ba2, const float* __restrict__ bc1,
    const float* __restrict__ Wc2, const float* __restrict__ bc2,
    float* __restrict__ out_h, float* __restrict__ out_c,
    float* __restrict__ out_logits, float* __restrict__ out_values) {
  __shared__ float    lds_bias[2 * 4 * DD];       // b_ih | b_hh (8 KB)
  __shared__ _Float16 lds_hf[16 * HF_STRIDE];     // h_new tile, f16
  __shared__ _Float16 lds_act[16 * ACT_STRIDE];   // all-expert hidden acts

  const int lane = threadIdx.x & 31;
  const int wave = threadIdx.x >> 5;
  const int row0 = blockIdx.x * 16;

  // Kick off async bias staging; completion consumed just before epilogue.
  {
    const int t4 = threadIdx.x * 4;                 // 256 threads x 4 floats
    async_cp_b128(lds_off_of(&lds_bias[t4]), b_ih + t4);
    async_cp_b128(lds_off_of(&lds_bias[4 * DD + t4]), b_hh + t4);
  }

  // ================= Phase 1: LSTM gates GEMM =================
  v8f acc[2][4];
  const v8f vzero = {0.f, 0.f, 0.f, 0.f, 0.f, 0.f, 0.f, 0.f};
#pragma unroll
  for (int t = 0; t < 2; ++t)
#pragma unroll
    for (int g = 0; g < 4; ++g) acc[t][g] = vzero;

#pragma unroll 1
  for (int ks = 0; ks < 8; ++ks) {
    const int k0 = ks * 32;
    v16h a_obs = load_a_f32(obs, DD, row0, k0, lane);
    v16h a_h   = load_a_f32(h,   DD, row0, k0, lane);
    if (ks < 7) {  // prefetch next k-chunk of the two true HBM streams
      const size_t pf = (size_t)(row0 + (lane & 15)) * DD + k0 + 32;
      __builtin_prefetch(obs + pf, 0, 3);
      __builtin_prefetch(h + pf, 0, 3);
    }
#pragma unroll
    for (int t = 0; t < 2; ++t) {
#pragma unroll
      for (int g = 0; g < 4; ++g) {
        const int ct = g * 16 + wave * 2 + t;  // gate-col tile in [0,64)
        v16h b1 = load_b_packed(pWih, ct, ks, lane);
        acc[t][g] = wmma16(a_obs, b1, acc[t][g]);
        v16h b2 = load_b_packed(pWhh, ct, ks, lane);
        acc[t][g] = wmma16(a_h, b2, acc[t][g]);
      }
    }
  }

  // Bias staging must be complete (own wave's asynccnt, then block barrier).
  wait_async0();
  __syncthreads();

  // ---- pointwise LSTM epilogue, all gates in registers ----
  {
    const int n = lane & 15, hi = lane >> 4;
#pragma unroll
    for (int t = 0; t < 2; ++t) {
      const int col = (wave * 2 + t) * 16 + n;
      const float bi = lds_bias[0 * DD + col] + lds_bias[4 * DD + 0 * DD + col];
      const float bf = lds_bias[1 * DD + col] + lds_bias[4 * DD + 1 * DD + col];
      const float bg = lds_bias[2 * DD + col] + lds_bias[4 * DD + 2 * DD + col];
      const float bo = lds_bias[3 * DD + col] + lds_bias[4 * DD + 3 * DD + col];
#pragma unroll
      for (int r = 0; r < 8; ++r) {
        const int mrow = r + hi * 8;
        const size_t off = (size_t)(row0 + mrow) * DD + col;
        const float iv = fast_sigmoid(acc[t][0][r] + bi);
        const float fv = fast_sigmoid(acc[t][1][r] + bf);
        const float gv = fast_tanh(acc[t][2][r] + bg);
        const float ov = fast_sigmoid(acc[t][3][r] + bo);
        const float cn = fv * c[off] + iv * gv;
        const float hn = ov * fast_tanh(cn);
        out_c[off] = cn;
        out_h[off] = hn;
        lds_hf[mrow * HF_STRIDE + col] = (_Float16)hn;
      }
    }
  }
  __syncthreads();

  // ================= Phase 2: actor layer-1, all 16 experts =================
  v8f acc1[8];
#pragma unroll
  for (int t = 0; t < 8; ++t) acc1[t] = vzero;
#pragma unroll 1
  for (int ks = 0; ks < 8; ++ks) {
    v16h a = load_a_lds(lds_hf, ks * 32, lane);
#pragma unroll
    for (int t = 0; t < 8; ++t)
      acc1[t] = wmma16(a, load_b_packed(pWa1, wave * 8 + t, ks, lane), acc1[t]);
  }
  {
    const int n = lane & 15, hi = lane >> 4;
#pragma unroll
    for (int t = 0; t < 8; ++t) {
      const int col0 = wave * 128 + t * 16;
      const float bias = ba1[(col0 >> 6) * HH + (col0 & 63) + n];
#pragma unroll
      for (int r = 0; r < 8; ++r)
        lds_act[(r + hi * 8) * ACT_STRIDE + col0 + n] =
            (_Float16)fast_tanh(acc1[t][r] + bias);
    }
  }
  __syncthreads();

  // ---- actor layer-2: per-row selected expert, 64x18 GEMV from LDS ----
  for (int idx = threadIdx.x; idx < 16 * AA; idx += 256) {
    const int r = idx / AA, a = idx % AA;
    const int e = idxs[row0 + r];
    float s = ba2[e * AA + a];
    const float* w2 = Wa2 + ((size_t)e * HH) * AA + a;
    const _Float16* hv = lds_act + r * ACT_STRIDE + e * HH;
#pragma unroll 8
    for (int k = 0; k < HH; ++k) s += (float)hv[k] * w2[(size_t)k * AA];
    out_logits[(size_t)(row0 + r) * AA + a] = s;
  }
  __syncthreads();

  // ================= Phase 3: critic layer-1, all 16 experts ================
#pragma unroll
  for (int t = 0; t < 8; ++t) acc1[t] = vzero;
#pragma unroll 1
  for (int ks = 0; ks < 8; ++ks) {
    v16h a = load_a_lds(lds_hf, ks * 32, lane);
#pragma unroll
    for (int t = 0; t < 8; ++t)
      acc1[t] = wmma16(a, load_b_packed(pWc1, wave * 8 + t, ks, lane), acc1[t]);
  }
  {
    const int n = lane & 15, hi = lane >> 4;
#pragma unroll
    for (int t = 0; t < 8; ++t) {
      const int col0 = wave * 128 + t * 16;
      const float bias = bc1[(col0 >> 6) * HH + (col0 & 63) + n];
#pragma unroll
      for (int r = 0; r < 8; ++r)
        lds_act[(r + hi * 8) * ACT_STRIDE + col0 + n] =
            (_Float16)fast_tanh(acc1[t][r] + bias);
    }
  }
  __syncthreads();

  // ---- critic layer-2: per-row selected expert, 64x1 dot from LDS ----
  if (threadIdx.x < 16) {
    const int r = threadIdx.x;
    const int e = idxs[row0 + r];
    float s = bc2[e];
    const float* w2 = Wc2 + (size_t)e * HH;
    const _Float16* hv = lds_act + r * ACT_STRIDE + e * HH;
#pragma unroll 8
    for (int k = 0; k < HH; ++k) s += (float)hv[k] * w2[k];
    out_values[row0 + r] = s;
  }
}

extern "C" void kernel_launch(void* const* d_in, const int* in_sizes, int n_in,
                              void* d_out, int out_size, void* d_ws, size_t ws_size,
                              hipStream_t stream) {
  (void)n_in; (void)out_size; (void)ws_size;
  const float* obs  = (const float*)d_in[0];
  const float* h    = (const float*)d_in[1];
  const float* c    = (const float*)d_in[2];
  const int*   idxs = (const int*)d_in[3];
  const float* W_ih = (const float*)d_in[4];
  const float* W_hh = (const float*)d_in[5];
  const float* b_ih = (const float*)d_in[6];
  const float* b_hh = (const float*)d_in[7];
  const float* Wa1  = (const float*)d_in[8];
  const float* ba1  = (const float*)d_in[9];
  const float* Wa2  = (const float*)d_in[10];
  const float* ba2  = (const float*)d_in[11];
  const float* Wc1  = (const float*)d_in[12];
  const float* bc1  = (const float*)d_in[13];
  const float* Wc2  = (const float*)d_in[14];
  const float* bc2  = (const float*)d_in[15];

  const int B = in_sizes[3];  // 65536

  // Workspace: 4 packed f16 fragment arrays, 512 KB each (2 MB total).
  _Float16* pWih = (_Float16*)d_ws;
  _Float16* pWhh = pWih + FRAG_HALVES;
  _Float16* pWa1 = pWhh + FRAG_HALVES;
  _Float16* pWc1 = pWa1 + FRAG_HALVES;

  const int prep_blocks = FRAG_HALVES / 256;
  pack_wT_kernel<<<prep_blocks, 256, 0, stream>>>(W_ih, pWih);
  pack_wT_kernel<<<prep_blocks, 256, 0, stream>>>(W_hh, pWhh);
  pack_wK_kernel<<<prep_blocks, 256, 0, stream>>>(Wa1, pWa1);
  pack_wK_kernel<<<prep_blocks, 256, 0, stream>>>(Wc1, pWc1);

  float* out_h      = (float*)d_out;
  float* out_c      = out_h + (size_t)B * DD;
  float* out_logits = out_c + (size_t)B * DD;
  float* out_values = out_logits + (size_t)B * AA;

  mlsh_fused_kernel<<<dim3(B / 16), dim3(256), 0, stream>>>(
      obs, h, c, idxs,
      (const v16h*)pWih, (const v16h*)pWhh, (const v16h*)pWa1, (const v16h*)pWc1,
      b_ih, b_hh, ba1, Wa2, ba2, bc1, Wc2, bc2,
      out_h, out_c, out_logits, out_values);
}
